// PositionAttentionModule_16260746183205
// MI455X (gfx1250) — compile-verified, matrix operand order checked by reading, not run on hardware
//
#include <hip/hip_runtime.h>

// ---------------------------------------------------------------------------
// PositionAttentionModule for MI455X (gfx1250), wave32 + WMMA f16->f32.
//   x  : (4, 512, 64, 64) f32      Wq,Wk: (64,512)  Wv: (256,512)  Wo: (512,256)
//   out: (4, 512, 64, 64) f32
// ~54 GFLOP vs ~67MB mandatory HBM traffic -> compute bound on the matrix
// units. All GEMMs via v_wmma_f32_16x16x32_f16; attention fused flash-style
// (4096x4096 scores never hit HBM); K/V chunks double-buffered in LDS, moved
// by the Tensor Data Mover (TENSORcnt) when available, else per-lane
// GLOBAL_LOAD_ASYNC_TO_LDS_B128 (ASYNCcnt), so global fetch overlaps WMMA.
// ---------------------------------------------------------------------------

typedef __attribute__((ext_vector_type(16))) _Float16 v16h;
typedef __attribute__((ext_vector_type(8)))  _Float16 v8h;
typedef __attribute__((ext_vector_type(8)))  float    v8f;

#define DEVI __device__ __forceinline__

#define BATCH 4
#define CIN   512
#define NTOK  4096     // 64*64
#define DQK   64
#define DV    256
#define CHUNK 32       // keys per flash iteration

#if defined(__gfx1250__) && __has_builtin(__builtin_amdgcn_tensor_load_to_lds)
#define HAVE_TDM 1
#else
#define HAVE_TDM 0
#endif

DEVI v8f wmma16(v16h a, v16h b, v8f c) {
  // emits v_wmma_f32_16x16x32_f16
  return __builtin_amdgcn_wmma_f32_16x16x32_f16(false, a, false, b,
                                                (short)0, c, false, false);
}

DEVI v8f zero8() { v8f z = {0.f,0.f,0.f,0.f,0.f,0.f,0.f,0.f}; return z; }

// Load one 16x32 (rows x K) operand tile in WMMA A/B register layout.
// p -> element (row0, k0) of a row-major matrix, leading dim ld (halves).
// ISA layout: lane holds row (lane&15); halves 0..7 = K {ks..ks+7},
// halves 8..15 = K {16+ks..16+ks+7}, ks = (lane>>4)*8. 16B-aligned loads.
DEVI v16h load_tile(const _Float16* p, int ld, int lane) {
  int m  = lane & 15;
  int ks = (lane >> 4) << 3;
  const _Float16* r = p + (size_t)m * ld + ks;
  v8h lo = *(const v8h*)(r);
  v8h hi = *(const v8h*)(r + 16);
  return __builtin_shufflevector(lo, hi, 0,1,2,3,4,5,6,7,8,9,10,11,12,13,14,15);
}

#if HAVE_TDM
typedef unsigned int u32x4 __attribute__((ext_vector_type(4)));
typedef int          i32x8 __attribute__((ext_vector_type(8)));
typedef int          i32x4 __attribute__((ext_vector_type(4)));

// Issue a 2-D TDM tile load (ISA ch.8 D# groups 0/1; groups 2/3 zero).
// Dims/strides in data_size units (f16 -> data_size code 1).  pad_enable
// inserts pad_amt DWORDs after every pad_int DWORDs into the LDS destination
// (hardware bank-conflict padding).  Codes: interval 2<<c DWORDs, amount c+1.
DEVI void tdm_load_2d(unsigned lds_addr, const void* gaddr,
                      unsigned tensor_d0, unsigned tensor_d1,
                      unsigned tile_d0, unsigned tile_d1,
                      unsigned stride0, unsigned pad_int_c, unsigned pad_amt_c) {
  unsigned long long ga = (unsigned long long)gaddr;
  u32x4 g0;
  g0[0] = 1u;                                       // count=1, user mode
  g0[1] = lds_addr;                                 // LDS byte address
  g0[2] = (unsigned)ga;                             // global_addr[31:0]
  g0[3] = (unsigned)(ga >> 32) | 0x80000000u;       // [56:32] | type=2
  i32x8 g1;
  g1[0] = (int)((1u << 16)            // data_size = 2 bytes
              | (1u << 20)            // pad_enable
              | (pad_int_c << 22) | (pad_amt_c << 25));
  g1[1] = (int)((tensor_d0 & 0xFFFFu) << 16);       // tensor_dim0[15:0]
  g1[2] = (int)((tensor_d0 >> 16) | ((tensor_d1 & 0xFFFFu) << 16));
  g1[3] = (int)((tensor_d1 >> 16) | (tile_d0 << 16));
  g1[4] = (int)tile_d1;                             // tile_dim1 (tile_dim2=0)
  g1[5] = (int)stride0;                             // tensor_dim0_stride
  g1[6] = 0;
  g1[7] = 0;
  i32x4 z4 = {0, 0, 0, 0};
  i32x8 z8 = {0, 0, 0, 0, 0, 0, 0, 0};
  // This toolchain's 6-arg form: (g0, g1, g2, g3, g4, cpol).
  __builtin_amdgcn_tensor_load_to_lds(g0, g1, z4, z4, z8, 0);
}
#else
// Async global->LDS copy of 16 bytes per lane (ISA 15.18.3 op 98, ASYNCcnt).
// VDST VGPR carries the LDS byte offset (low 32 bits of the generic pointer).
DEVI void async_copy_b128(const _Float16* g, _Float16* l) {
  unsigned lds = (unsigned)(size_t)(void*)l;
  asm volatile("global_load_async_to_lds_b128 %0, %1, off"
               :: "v"(lds), "v"(g) : "memory");
}
#endif

// ---------------------------------------------------------------------------
// Stage 0a: fp32 -> f16 copy (weights keep their (d,c)/(cout,d) layout, which
// is exactly the contiguous-K "B-tile" layout WMMA loads want).
// ---------------------------------------------------------------------------
__global__ void cvt_f16_kernel(const float* __restrict__ src,
                               _Float16* __restrict__ dst, int n) {
  int i = blockIdx.x * blockDim.x + threadIdx.x;
  if (i < n) dst[i] = (_Float16)src[i];
}

// ---------------------------------------------------------------------------
// Stage 0b: x (b, c=512, n=4096) f32 -> xh (b, n, c) f16 (LDS-tiled transpose)
// so activation rows are contiguous-K WMMA A tiles.
// ---------------------------------------------------------------------------
__global__ __launch_bounds__(256) void xpose_kernel(const float* __restrict__ x,
                                                    _Float16* __restrict__ xh) {
  __shared__ float tile[32][33];
  int b  = blockIdx.z;
  int c0 = blockIdx.y * 32;
  int n0 = blockIdx.x * 32;
  int tx = threadIdx.x, ty = threadIdx.y;          // (32, 8)
  const float* xp = x + ((size_t)b * CIN + c0) * NTOK + n0;
  #pragma unroll
  for (int i = 0; i < 4; ++i)
    tile[ty + 8 * i][tx] = xp[(size_t)(ty + 8 * i) * NTOK + tx];
  __syncthreads();
  _Float16* op = xh + ((size_t)b * NTOK + n0) * CIN + c0;
  #pragma unroll
  for (int i = 0; i < 4; ++i)
    op[(size_t)(ty + 8 * i) * CIN + tx] = (_Float16)tile[tx][ty + 8 * i];
}

// ---------------------------------------------------------------------------
// Stage 1: QKV projection.  Q,K: (n,64) row-major.  V is stored TRANSPOSED,
// (d_v, n) per batch, so the attention kernel's V staging is a contiguous
// tile copy (the transpose is paid once here instead of 64x in stage 2).
// Wave = 16 activation rows x one 64-col tile group; K-loop over c=512.
// tile group: 0 -> Q, 1 -> K, 2..5 -> V cols [ (tg-2)*64 .. +64 ).
// ---------------------------------------------------------------------------
__global__ __launch_bounds__(128) void qkv_kernel(
    const _Float16* __restrict__ xh,
    const _Float16* __restrict__ Wqh, const _Float16* __restrict__ Wkh,
    const _Float16* __restrict__ Wvh,
    _Float16* __restrict__ Qh, _Float16* __restrict__ Kh,
    _Float16* __restrict__ Vg) {
  int tid  = blockIdx.x * blockDim.x + threadIdx.x;
  int w    = tid >> 5;
  int lane = tid & 31;
  int tg = w % 6;
  int rb = w / 6;                       // [0, 4*256)
  int batch = rb >> 8;
  int nb    = (rb & 255) << 4;          // 16-row block

  const _Float16* A = xh + ((size_t)batch * NTOK + nb) * CIN;
  const _Float16* W;
  int co;
  if (tg == 0)      { W = Wqh; co = 0; }
  else if (tg == 1) { W = Wkh; co = 0; }
  else              { W = Wvh; co = (tg - 2) * 64; }

  v8f acc[4];
  #pragma unroll
  for (int j = 0; j < 4; ++j) acc[j] = zero8();

  #pragma unroll 4
  for (int kc = 0; kc < CIN; kc += 32) {
    v16h a = load_tile(A + kc, CIN, lane);
    #pragma unroll
    for (int j = 0; j < 4; ++j) {
      v16h b = load_tile(W + (size_t)(co + 16 * j) * CIN + kc, CIN, lane);
      acc[j] = wmma16(a, b, acc[j]);
    }
  }
  int hi = lane >> 4, nn = lane & 15;
  if (tg < 2) {
    _Float16* out = (tg == 0 ? Qh : Kh) + (size_t)batch * NTOK * DQK;
    #pragma unroll
    for (int j = 0; j < 4; ++j)
      #pragma unroll
      for (int r = 0; r < 8; ++r)
        out[(size_t)(nb + r + 8 * hi) * DQK + 16 * j + nn] = (_Float16)acc[j][r];
  } else {
    // transposed store: Vg[b][dv][n]
    _Float16* out = Vg + (size_t)batch * DV * NTOK;
    #pragma unroll
    for (int j = 0; j < 4; ++j)
      #pragma unroll
      for (int r = 0; r < 8; ++r)
        out[(size_t)(co + 16 * j + nn) * NTOK + nb + r + 8 * hi] =
            (_Float16)acc[j][r];
  }
}

// ---------------------------------------------------------------------------
// Stage 2: flash attention.  Block = 4 waves x 16 query rows = 64 rows.
// Per chunk of 32 keys: K (32x64) and V^T (256x32) staged in double-buffered
// LDS one chunk ahead.  TDM path: wave 0 issues two tensor_load_to_lds per
// chunk (hardware LDS padding), s_wait_tensorcnt 2 retires the current chunk
// while the next is in flight.  Fallback: 10 per-lane async-to-LDS b128 ops
// per thread, s_wait_asynccnt 10.  Then S = Q K^T (4 WMMAs), online softmax
// via xor-shuffle row reductions (rows live in 16-lane halves under wave32),
// P bounced via LDS C->A layout, O(16x256) += P V (16 WMMAs) in 128 VGPRs.
// ---------------------------------------------------------------------------
__global__ __launch_bounds__(128) void attn_kernel(
    const _Float16* __restrict__ Qh, const _Float16* __restrict__ Kh,
    const _Float16* __restrict__ Vg, _Float16* __restrict__ Oh) {
  __shared__ alignas(16) _Float16 Kc[2][CHUNK][72];   // 32 x (64 + 8 pad)
  __shared__ alignas(16) _Float16 Vt[2][DV][40];      // (dv, 32 keys + 8 pad)
  __shared__ alignas(16) _Float16 Pb[4][16][40];      // per-wave P bounce

  int tid  = threadIdx.x;
  int lane = tid & 31;
  int wv   = tid >> 5;
  int batch = blockIdx.x >> 6;
  int rb    = blockIdx.x & 63;
  int nbase = rb * 64 + wv * 16;

  const size_t qoff = (size_t)batch * NTOK * DQK;
  const size_t voff = (size_t)batch * NTOK * DV;
  const _Float16* Kb = Kh + qoff;
  const _Float16* Vb = Vg + voff;

  // Persistent Q (16x64) as two A tiles.
  v16h qa0 = load_tile(Qh + qoff + (size_t)nbase * DQK + 0,  DQK, lane);
  v16h qa1 = load_tile(Qh + qoff + (size_t)nbase * DQK + 32, DQK, lane);

  v8f o[16];
  #pragma unroll
  for (int t = 0; t < 16; ++t) o[t] = zero8();
  float rmax[8], rsum[8];
  #pragma unroll
  for (int r = 0; r < 8; ++r) { rmax[r] = -1e30f; rsum[r] = 0.f; }

#if HAVE_TDM
  // One wave drives the Tensor Data Mover for the whole workgroup.
  auto stage_chunk = [&](int kb, int buf) {
    if (wv == 0) {
      // K tile: 32 rows x 64 halves; rows padded 32+4 DWORDs -> stride 72.
      tdm_load_2d((unsigned)(size_t)(void*)&Kc[buf][0][0],
                  Kb + (size_t)kb * DQK,
                  /*tensor_d0=*/DQK, /*tensor_d1=*/NTOK,
                  /*tile_d0=*/DQK, /*tile_d1=*/CHUNK,
                  /*stride0=*/DQK, /*pad: 32 DW*/4, /*+4 DW*/3);
      // V^T tile: 256 rows x 32 halves; rows padded 16+4 DWORDs -> stride 40.
      tdm_load_2d((unsigned)(size_t)(void*)&Vt[buf][0][0],
                  Vb + kb,
                  /*tensor_d0=*/NTOK, /*tensor_d1=*/DV,
                  /*tile_d0=*/CHUNK, /*tile_d1=*/DV,
                  /*stride0=*/NTOK, /*pad: 16 DW*/3, /*+4 DW*/3);
    }
  };
#else
  // Per-lane async staging: K 32x64 (256 b128 segs) + V^T 256x32 (1024 segs)
  // over 128 threads = 10 async ops per thread.
  auto stage_chunk = [&](int kb, int buf) {
    _Float16 (*kc)[72] = Kc[buf];
    _Float16 (*vt)[40] = Vt[buf];
    #pragma unroll
    for (int i = 0; i < 2; ++i) {
      int id = tid + 128 * i, key = id >> 3, seg = id & 7;
      async_copy_b128(Kb + (size_t)(kb + key) * DQK + seg * 8,
                      &kc[key][seg * 8]);
    }
    #pragma unroll
    for (int i = 0; i < 8; ++i) {
      int id = tid + 128 * i, dv = id >> 2, seg = id & 3;
      async_copy_b128(Vb + (size_t)dv * NTOK + kb + seg * 8,
                      &vt[dv][seg * 8]);
    }
  };
#endif

  const int NCH = NTOK / CHUNK;
  stage_chunk(0, 0);

  for (int j = 0; j < NCH; ++j) {
    int cur = j & 1;
    __syncthreads();   // all waves done computing chunk j-1 -> buf cur^1 free
    if (j + 1 < NCH) stage_chunk((j + 1) * CHUNK, cur ^ 1);
#if HAVE_TDM
    if (wv == 0) {
      if (j + 1 < NCH) __builtin_amdgcn_s_wait_tensorcnt(2);  // chunk j landed
      else             __builtin_amdgcn_s_wait_tensorcnt(0);
    }
#else
    if (j + 1 < NCH) asm volatile("s_wait_asynccnt 10" ::: "memory");
    else             asm volatile("s_wait_asynccnt 0"  ::: "memory");
#endif
    __syncthreads();   // chunk j visible to all waves

    // S = Q K^T : two 16x16 tiles over 32 keys.
    v8f s0 = zero8(), s1 = zero8();
    {
      v16h b0 = load_tile(&Kc[cur][0][0],   72, lane);
      v16h b1 = load_tile(&Kc[cur][0][32],  72, lane);
      s0 = wmma16(qa0, b0, s0);
      s0 = wmma16(qa1, b1, s0);
      v16h b2 = load_tile(&Kc[cur][16][0],  72, lane);
      v16h b3 = load_tile(&Kc[cur][16][32], 72, lane);
      s1 = wmma16(qa0, b2, s1);
      s1 = wmma16(qa1, b3, s1);
    }

    // Online softmax. Accumulator row M = r + 8*(lane>>4); row reductions are
    // xor-shuffles within 16-lane halves.
    float p0[8], p1[8], sc[8];
    #pragma unroll
    for (int r = 0; r < 8; ++r) {
      float v = fmaxf(s0[r], s1[r]);
      v = fmaxf(v, __shfl_xor(v, 1));
      v = fmaxf(v, __shfl_xor(v, 2));
      v = fmaxf(v, __shfl_xor(v, 4));
      v = fmaxf(v, __shfl_xor(v, 8));
      float mnew = fmaxf(rmax[r], v);
      sc[r] = __expf(rmax[r] - mnew);
      rmax[r] = mnew;
      p0[r] = __expf(s0[r] - mnew);
      p1[r] = __expf(s1[r] - mnew);
      float su = p0[r] + p1[r];
      su += __shfl_xor(su, 1);
      su += __shfl_xor(su, 2);
      su += __shfl_xor(su, 4);
      su += __shfl_xor(su, 8);
      rsum[r] = rsum[r] * sc[r] + su;
    }
    #pragma unroll
    for (int t = 0; t < 16; ++t)
      #pragma unroll
      for (int r = 0; r < 8; ++r) o[t][r] *= sc[r];

    // P: C/D layout -> LDS -> A layout (wave-private region; LDS is in-order
    // per wave, s_wait_dscnt makes the hand-off explicit).
    {
      int hi = lane >> 4, nn = lane & 15;
      #pragma unroll
      for (int r = 0; r < 8; ++r) {
        Pb[wv][r + 8 * hi][nn]      = (_Float16)p0[r];
        Pb[wv][r + 8 * hi][16 + nn] = (_Float16)p1[r];
      }
    }
    asm volatile("s_wait_dscnt 0" ::: "memory");
    v16h pa = load_tile(&Pb[wv][0][0], 40, lane);

    // O(16x256) += P(16x32) * V(32x256)
    #pragma unroll
    for (int t = 0; t < 16; ++t) {
      v16h bv = load_tile(&Vt[cur][t * 16][0], 40, lane);
      o[t] = wmma16(pa, bv, o[t]);
    }
  }

  // Normalize and store O (n, 256) f16.
  float inv[8];
  #pragma unroll
  for (int r = 0; r < 8; ++r) inv[r] = 1.0f / rsum[r];
  int hi = lane >> 4, nn = lane & 15;
  #pragma unroll
  for (int t = 0; t < 16; ++t)
    #pragma unroll
    for (int r = 0; r < 8; ++r)
      Oh[voff + (size_t)(nbase + r + 8 * hi) * DV + t * 16 + nn] =
          (_Float16)(o[t][r] * inv[r]);
}

// ---------------------------------------------------------------------------
// Stage 3: out(b, 512, 4096) f32 = Wo(512,256) @ O^T(256,4096).
// A = Wo rows (cout, K=d contiguous); B tile reads O rows (n, d contiguous).
// Wave = one 16-row cout tile x 4 n-tiles.
// ---------------------------------------------------------------------------
__global__ __launch_bounds__(256) void oproj_kernel(
    const _Float16* __restrict__ Woh, const _Float16* __restrict__ Oh,
    float* __restrict__ out) {
  int tid  = blockIdx.x * blockDim.x + threadIdx.x;
  int w    = tid >> 5;
  int lane = tid & 31;
  int ng    = w & 63;           // 64 groups of 64 cols
  int ct    = (w >> 6) & 31;    // 32 cout tiles
  int batch = w >> 11;
  int nb = ng * 64;
  int cb = ct * 16;
  const _Float16* Ob = Oh + (size_t)batch * NTOK * DV;

  v8f acc[4];
  #pragma unroll
  for (int j = 0; j < 4; ++j) acc[j] = zero8();

  #pragma unroll
  for (int kc = 0; kc < DV; kc += 32) {
    v16h a = load_tile(Woh + (size_t)cb * DV + kc, DV, lane);
    #pragma unroll
    for (int j = 0; j < 4; ++j) {
      v16h b = load_tile(Ob + (size_t)(nb + 16 * j) * DV + kc, DV, lane);
      acc[j] = wmma16(a, b, acc[j]);
    }
  }
  int hi = lane >> 4, nn = lane & 15;
  #pragma unroll
  for (int j = 0; j < 4; ++j)
    #pragma unroll
    for (int r = 0; r < 8; ++r)
      out[((size_t)batch * CIN + cb + r + 8 * hi) * NTOK + nb + 16 * j + nn] =
          acc[j][r];
}

// ---------------------------------------------------------------------------
extern "C" void kernel_launch(void* const* d_in, const int* in_sizes, int n_in,
                              void* d_out, int out_size, void* d_ws, size_t ws_size,
                              hipStream_t stream) {
  const float* x  = (const float*)d_in[0];
  const float* Wq = (const float*)d_in[1];
  const float* Wk = (const float*)d_in[2];
  const float* Wv = (const float*)d_in[3];
  const float* Wo = (const float*)d_in[4];
  float* out = (float*)d_out;

  _Float16* ws = (_Float16*)d_ws;
  size_t off = 0;
  _Float16* xh  = ws + off; off += (size_t)BATCH * NTOK * CIN;   // 8.4M halves
  _Float16* Wqh = ws + off; off += (size_t)DQK * CIN;
  _Float16* Wkh = ws + off; off += (size_t)DQK * CIN;
  _Float16* Wvh = ws + off; off += (size_t)DV * CIN;
  _Float16* Woh = ws + off; off += (size_t)CIN * DV;
  _Float16* Qh  = ws + off; off += (size_t)BATCH * NTOK * DQK;
  _Float16* Kh  = ws + off; off += (size_t)BATCH * NTOK * DQK;
  _Float16* Vg  = ws + off; off += (size_t)BATCH * NTOK * DV;    // (b, dv, n)
  _Float16* Oh  = ws + off; off += (size_t)BATCH * NTOK * DV;    // (b, n, dv)

  // Stage 0: precision conversion + activation transpose.
  cvt_f16_kernel<<<(DQK * CIN + 255) / 256, 256, 0, stream>>>(Wq, Wqh, DQK * CIN);
  cvt_f16_kernel<<<(DQK * CIN + 255) / 256, 256, 0, stream>>>(Wk, Wkh, DQK * CIN);
  cvt_f16_kernel<<<(DV * CIN + 255) / 256, 256, 0, stream>>>(Wv, Wvh, DV * CIN);
  cvt_f16_kernel<<<(CIN * DV + 255) / 256, 256, 0, stream>>>(Wo, Woh, CIN * DV);
  dim3 tg(NTOK / 32, CIN / 32, BATCH);
  xpose_kernel<<<tg, dim3(32, 8), 0, stream>>>(x, xh);

  // Stage 1: QKV projections (6144 waves).
  qkv_kernel<<<(BATCH * 256 * 6 * 32) / 128, 128, 0, stream>>>(
      xh, Wqh, Wkh, Wvh, Qh, Kh, Vg);

  // Stage 2: fused flash attention (256 blocks x 4 waves).
  attn_kernel<<<BATCH * (NTOK / 64), 128, 0, stream>>>(Qh, Kh, Vg, Oh);

  // Stage 3: output projection (8192 waves).
  oproj_kernel<<<(BATCH * 32 * 64 * 32) / 256, 256, 0, stream>>>(Woh, Oh, out);
}
